// PolicyNet_84920093376584
// MI455X (gfx1250) — matrix operand order repaired and checked by reading, not compiled, not run on hardware
//
#include <hip/hip_runtime.h>
#include <hip/hip_bf16.h>
#include <math.h>

// ---------------------------------------------------------------------------
// CDNA5 / gfx1250 implementation of the GATv2 policy network.
//   - All GEMMs run on v_wmma_f32_16x16x32_f16 (f16 in, f32 accumulate).
//   - Weights transposed+converted to f16 so B fragments are contiguous
//     32B loads (Wt[k*Nc+n]); A fragments are two 16B vector loads per lane
//     with row clamping (no exec-predicated scalar loads).
//   - GEMM: 2 M-tiles x NT N-tiles per wave => each B fragment feeds two
//     WMMAs (24B loaded per WMMA); scalar A pointers (not arrays) so clang
//     keeps global address-space inference -> global_load_b128, not flat.
//   - Segment softmax via float atomicMax (ordered-int trick) + atomicAdd.
//   - Edge MLP processed in chunks sized from ws_size (deterministic).
// ---------------------------------------------------------------------------

typedef __attribute__((ext_vector_type(16))) _Float16 v16h;
typedef __attribute__((ext_vector_type(8)))  _Float16 v8h;
typedef __attribute__((ext_vector_type(8)))  float    v8f;

#define HEADS 4
#define NGOODS 16

// ---------------------------- small helpers --------------------------------

__device__ __forceinline__ void atomicMaxF(float* addr, float v) {
    // works for IEEE754 with -FLT_MAX init: positives via signed-int max,
    // negatives via unsigned min (bit pattern order reversed for negatives)
    if (v >= 0.0f) atomicMax((int*)addr, __float_as_int(v));
    else           atomicMin((unsigned int*)addr, __float_as_uint(v));
}

__global__ void fill_f32_kernel(float* p, float v, int n) {
    int t = blockIdx.x * blockDim.x + threadIdx.x;
    if (t < n) p[t] = v;
}

__global__ void zero_scalar_kernel(float* p) {
    if (blockIdx.x == 0 && threadIdx.x == 0) p[0] = 0.0f;
}

// h = concat(x[N,63], next_demand[N,1]) -> f16 [N,64]
__global__ void build_h16_kernel(const float* x, const float* nd, _Float16* h16, int N) {
    int t = blockIdx.x * blockDim.x + threadIdx.x;
    if (t >= N * 64) return;
    int n = t >> 6, c = t & 63;
    float v = (c < 63) ? x[n * 63 + c] : nd[n];
    h16[t] = (_Float16)v;
}

// W [Nc,K] f32 row-major  ->  Wt [K,Nc] f16  (so B(k,n)=Wt[k*Nc+n] contiguous in n)
__global__ void wconv_kernel(const float* W, _Float16* Wt, int Nc, int K) {
    int t = blockIdx.x * blockDim.x + threadIdx.x;
    if (t >= K * Nc) return;
    int k = t / Nc, n = t % Nc;
    Wt[t] = (_Float16)W[n * K + k];
}

__device__ __forceinline__ v16h load_afrag(const _Float16* p) {
    v8h alo = *(const v8h*)(p);            // 16B-aligned vector loads
    v8h ahi = *(const v8h*)(p + 16);
    return __builtin_shufflevector(alo, ahi, 0, 1, 2, 3, 4, 5, 6, 7,
                                   8, 9, 10, 11, 12, 13, 14, 15);
}

// ---------------------------- WMMA GEMM ------------------------------------
// out[M,Nc] = act(A16[M,K] @ Wt16[K,Nc] + bias)
// 2 M-tiles x NT N-tiles per wave; B fragments reused across M-tiles.
// A-frag layout (ISA 7.12.2, 16-bit A 16x32): lane = hi*16 + r holds row r,
//   halves j=0..7 -> K = 8*hi + j, halves j=8..15 -> K = 16 + 8*hi + (j-8)
//   => two contiguous 16B chunks at half-offsets 8*hi and 16+8*hi.
// B-frag (32x16): lane holds column n = lane&15, half j -> same K as A (8*hi+j / 16+8*hi+(j-8)).
// C/D (16x16 f32): lane -> n = lane&15, vgpr i -> m = i + 8*hi.
template <int NT>
__global__ __launch_bounds__(256)
void wmma_gemm_kernel(const _Float16* __restrict__ A,
                      const _Float16* __restrict__ Wt,
                      const float* __restrict__ bias,
                      float* __restrict__ outF,
                      _Float16* __restrict__ outH,
                      int M, int K, int Nc, int relu) {
    int wid = blockIdx.x * (blockDim.x >> 5) + (threadIdx.x >> 5);
    int mgrps = (M + 31) >> 5;
    int ngrps = Nc / (16 * NT);
    if (wid >= mgrps * ngrps) return;
    int mg = wid / ngrps, ng = wid % ngrps;

    int lane = threadIdx.x & 31;
    int r = lane & 15;
    int hi = lane >> 4;
    int nbase = ng * (16 * NT);
    int ncol = lane & 15;

    v8f acc0[NT], acc1[NT];
#pragma unroll
    for (int t = 0; t < NT; ++t) {
        float bv = bias ? bias[nbase + t * 16 + ncol] : 0.0f;
#pragma unroll
        for (int i = 0; i < 8; ++i) { acc0[t][i] = bv; acc1[t][i] = bv; }
    }

    // clamp: out-of-range rows read valid memory; their outputs are never stored
    int row0 = mg * 32 + r;
    int row1 = row0 + 16;
    int rowc0 = row0 < M ? row0 : (M - 1);
    int rowc1 = row1 < M ? row1 : (M - 1);
    const _Float16* ap0 = A + (size_t)rowc0 * K + 8 * hi;
    const _Float16* ap1 = A + (size_t)rowc1 * K + 8 * hi;
    const _Float16* bp = Wt + (size_t)(8 * hi) * Nc + nbase + ncol;
    const size_t bstep = (size_t)32 * Nc;          // halves per K-step

    for (int kb = 0; kb < K; kb += 32) {
        v16h a0 = load_afrag(ap0);
        v16h a1 = load_afrag(ap1);
        ap0 += 32;
        ap1 += 32;
#pragma unroll
        for (int t = 0; t < NT; ++t) {
            v16h b;
#pragma unroll
            for (int s = 0; s < 16; ++s) b[s] = bp[t * 16 + (size_t)(s < 8 ? s : s + 8) * Nc];
            acc0[t] = __builtin_amdgcn_wmma_f32_16x16x32_f16(
                false, a0, false, b, (short)0, acc0[t], false, false);
            acc1[t] = __builtin_amdgcn_wmma_f32_16x16x32_f16(
                false, a1, false, b, (short)0, acc1[t], false, false);
        }
        bp += bstep;
    }

#pragma unroll
    for (int t = 0; t < NT; ++t)
#pragma unroll
        for (int i = 0; i < 8; ++i) {
            int g0 = mg * 32 + i + 8 * hi;
            int g1 = g0 + 16;
            int col = nbase + t * 16 + ncol;
            if (g0 < M) {
                float v = acc0[t][i];
                if (relu) v = v > 0.0f ? v : 0.0f;
                size_t o = (size_t)g0 * Nc + col;
                if (outF) outF[o] = v;
                if (outH) outH[o] = (_Float16)v;
            }
            if (g1 < M) {
                float v = acc1[t][i];
                if (relu) v = v > 0.0f ? v : 0.0f;
                size_t o = (size_t)g1 * Nc + col;
                if (outF) outF[o] = v;
                if (outH) outH[o] = (_Float16)v;
            }
        }
}

// ---------------------------- GATv2 kernels --------------------------------
// edges e in [0,E0): src=esrc[e], dst=edst[e]; e in [E0,Etot): self loop e-E0.

template <int H, int C>
__global__ void gat_score_kernel(const float* __restrict__ xl, const float* __restrict__ xr,
                                 const float* __restrict__ att,
                                 const int* __restrict__ esrc, const int* __restrict__ edst,
                                 int E0, int Etot, float* __restrict__ score,
                                 float* __restrict__ mbuf) {
    int t = blockIdx.x * blockDim.x + threadIdx.x;
    if (t >= Etot * H) return;
    int e = t / H, h = t % H;
    int src = (e < E0) ? esrc[e] : (e - E0);
    int dst = (e < E0) ? edst[e] : (e - E0);
    const float* xls = xl + (size_t)src * H * C + h * C;
    const float* xrd = xr + (size_t)dst * H * C + h * C;
    const float* ah = att + h * C;
    float s = 0.0f;
#pragma unroll 8
    for (int c = 0; c < C; ++c) {
        float v = xls[c] + xrd[c];
        v = v > 0.0f ? v : 0.2f * v;   // leaky_relu(0.2)
        s += ah[c] * v;
    }
    score[t] = s;
    atomicMaxF(&mbuf[dst * H + h], s);
}

template <int H>
__global__ void gat_expsum_kernel(float* __restrict__ score, const float* __restrict__ mbuf,
                                  const int* __restrict__ esrc, const int* __restrict__ edst,
                                  int E0, int Etot, float* __restrict__ den) {
    int t = blockIdx.x * blockDim.x + threadIdx.x;
    if (t >= Etot * H) return;
    int e = t / H, h = t % H;
    int dst = (e < E0) ? edst[e] : (e - E0);
    float ex = __expf(score[t] - mbuf[dst * H + h]);
    score[t] = ex;                       // overwrite score with exp in-place
    atomicAdd(&den[dst * H + h], ex);
}

template <int H, int C>
__global__ void gat_aggr_kernel(const float* __restrict__ ex, const float* __restrict__ den,
                                const float* __restrict__ xl,
                                const int* __restrict__ esrc, const int* __restrict__ edst,
                                int E0, int Etot, float* __restrict__ agg) {
    int t = blockIdx.x * blockDim.x + threadIdx.x;
    if (t >= Etot * H) return;
    int e = t / H, h = t % H;
    int src = (e < E0) ? esrc[e] : (e - E0);
    int dst = (e < E0) ? edst[e] : (e - E0);
    float alpha = ex[t] / (den[dst * H + h] + 1e-16f);
    const float* xls = xl + (size_t)src * H * C + h * C;
    float* o = agg + (size_t)dst * H * C + h * C;
#pragma unroll 8
    for (int c = 0; c < C; ++c) atomicAdd(&o[c], alpha * xls[c]);
}

// y16 = leaky_relu(agg + bias)  (H==1 "mean" is identity)
__global__ void gat_finish_kernel(const float* __restrict__ agg, const float* __restrict__ bias,
                                  _Float16* __restrict__ y16, int N, int HC) {
    int t = blockIdx.x * blockDim.x + threadIdx.x;
    if (t >= N * HC) return;
    int d = t % HC;
    float v = agg[t] + bias[d];
    v = v > 0.0f ? v : 0.2f * v;
    y16[t] = (_Float16)v;
}

// s16[n] = [xf2_16[n] (64) | xb4_16[n] (64)]
__global__ void concat_s_kernel(const _Float16* xf, const _Float16* xb, _Float16* s16, int N) {
    int t = blockIdx.x * blockDim.x + threadIdx.x;
    if (t >= N * 128) return;
    int n = t >> 7, c = t & 127;
    s16[t] = (c < 64) ? xf[n * 64 + c] : xb[n * 64 + (c - 64)];
}

// se16[i] = [s16[src(e)] | s16[dst(e)]], e = e0+i (original edges, no self loops)
__global__ void gather_se_kernel(const _Float16* __restrict__ s16, const int* __restrict__ ei,
                                 int E, int e0, int cm, _Float16* __restrict__ se16) {
    int t = blockIdx.x * blockDim.x + threadIdx.x;
    if (t >= cm * 256) return;
    int i = t >> 8, j = t & 255;
    int e = e0 + i;
    int node = (j < 128) ? ei[e] : ei[E + e];
    se16[t] = s16[(size_t)node * 128 + (j & 127)];
}

// tanh-squashed action + log-prob reduction
__global__ __launch_bounds__(256)
void mlp_head_kernel(const float* __restrict__ mu, const float* __restrict__ stdpre,
                     const float* __restrict__ eps, int e0, int cm,
                     float* __restrict__ outA, float* __restrict__ outS) {
    int t = blockIdx.x * blockDim.x + threadIdx.x;
    float contrib = 0.0f;
    int tot = cm * NGOODS;
    if (t < tot) {
        int i = t / NGOODS, g = t % NGOODS;
        float m  = mu[i * NGOODS + g];
        float sp = stdpre[i * NGOODS + g];
        float sd = (sp > 20.0f) ? sp : log1pf(__expf(sp));   // softplus
        float ev = eps[(size_t)(e0 + i) * NGOODS + g];
        float action = m + sd * ev;
        float ra = tanhf(action);
        outA[(size_t)(e0 + i) * NGOODS + g] = ra;
        // (action-mu)/std == eps exactly
        float lp = -0.5f * ev * ev - __logf(sd) - 0.91893853320467274f;
        float corr = __logf(1.0f - ra * ra + 1e-7f);
        contrib = lp - corr;
    }
    __shared__ float red[256];
    red[threadIdx.x] = contrib;
    __syncthreads();
    for (int s = 128; s > 0; s >>= 1) {
        if (threadIdx.x < s) red[threadIdx.x] += red[threadIdx.x + s];
        __syncthreads();
    }
    if (threadIdx.x == 0) atomicAdd(outS, red[0]);
}

// ---------------------------- host side ------------------------------------

extern "C" void kernel_launch(void* const* d_in, const int* in_sizes, int n_in,
                              void* d_out, int out_size, void* d_ws, size_t ws_size,
                              hipStream_t stream) {
    const float* x    = (const float*)d_in[0];
    const float* nd   = (const float*)d_in[1];
    const int*   ei   = (const int*)d_in[2];     // [2,E]
    const float* eps  = (const float*)d_in[3];
    const float* Wl3 = (const float*)d_in[4],  *Wr3 = (const float*)d_in[5];
    const float* att3 = (const float*)d_in[6], *b3  = (const float*)d_in[7];
    const float* Wl4 = (const float*)d_in[8],  *Wr4 = (const float*)d_in[9];
    const float* att4 = (const float*)d_in[10], *b4 = (const float*)d_in[11];
    const float* Wl1 = (const float*)d_in[12], *Wr1 = (const float*)d_in[13];
    const float* att1 = (const float*)d_in[14], *b1 = (const float*)d_in[15];
    const float* Wl2 = (const float*)d_in[16], *Wr2 = (const float*)d_in[17];
    const float* att2 = (const float*)d_in[18], *b2 = (const float*)d_in[19];
    const float* Wfc  = (const float*)d_in[20], *bfc  = (const float*)d_in[21];
    const float* Wfc1 = (const float*)d_in[22], *bfc1 = (const float*)d_in[23];
    const float* Wmu  = (const float*)d_in[24], *bmu  = (const float*)d_in[25];
    const float* Wstd = (const float*)d_in[26], *bstd = (const float*)d_in[27];

    const int N = in_sizes[0] / 63;
    const int E = in_sizes[2] / 2;
    const int Etot = E + N;                 // + self loops
    float* out = (float*)d_out;
    float* out_scalar = out + (size_t)E * NGOODS;

    // -- workspace allocator --
    size_t off = 0;
    auto alloc = [&](size_t bytes) -> char* {
        char* p = (char*)d_ws + off;
        off = (off + bytes + 255) & ~(size_t)255;
        return p;
    };
    // f16 transposed weights
    _Float16* wl3t = (_Float16*)alloc(64 * 128 * 2);
    _Float16* wr3t = (_Float16*)alloc(64 * 128 * 2);
    _Float16* wl4t = (_Float16*)alloc(128 * 64 * 2);
    _Float16* wr4t = (_Float16*)alloc(128 * 64 * 2);
    _Float16* wl1t = (_Float16*)alloc(64 * 128 * 2);
    _Float16* wr1t = (_Float16*)alloc(64 * 128 * 2);
    _Float16* wl2t = (_Float16*)alloc(128 * 64 * 2);
    _Float16* wr2t = (_Float16*)alloc(128 * 64 * 2);
    _Float16* wfct  = (_Float16*)alloc((size_t)256 * 512 * 2);
    _Float16* wfc1t = (_Float16*)alloc((size_t)512 * 256 * 2);
    _Float16* wmut  = (_Float16*)alloc(256 * 16 * 2);
    _Float16* wstdt = (_Float16*)alloc(256 * 16 * 2);
    // node buffers
    _Float16* h16    = (_Float16*)alloc((size_t)N * 64 * 2);
    float*    xl     = (float*)alloc((size_t)N * 128 * 4);
    float*    xr     = (float*)alloc((size_t)N * 128 * 4);
    float*    mbuf   = (float*)alloc((size_t)N * HEADS * 4);
    float*    den    = (float*)alloc((size_t)N * HEADS * 4);
    float*    score  = (float*)alloc((size_t)Etot * HEADS * 4);
    float*    agg    = (float*)alloc((size_t)N * 128 * 4);
    _Float16* xb3_16 = (_Float16*)alloc((size_t)N * 128 * 2);
    _Float16* xb4_16 = (_Float16*)alloc((size_t)N * 64 * 2);
    _Float16* xf1_16 = (_Float16*)alloc((size_t)N * 128 * 2);
    _Float16* xf2_16 = (_Float16*)alloc((size_t)N * 64 * 2);
    _Float16* s16    = (_Float16*)alloc((size_t)N * 128 * 2);

    // chunk size for the edge MLP from remaining workspace (2176 B/edge)
    size_t rem = (ws_size > off) ? (ws_size - off) : 0;
    int CH = (int)(rem / 2176);
    if (CH > E) CH = E;
    CH &= ~31;
    if (CH < 32) CH = 32;
    _Float16* se16  = (_Float16*)alloc((size_t)CH * 256 * 2);
    _Float16* h1_16 = (_Float16*)alloc((size_t)CH * 512 * 2);
    _Float16* h2_16 = (_Float16*)alloc((size_t)CH * 256 * 2);
    float*    muB   = (float*)alloc((size_t)CH * 16 * 4);
    float*    spB   = (float*)alloc((size_t)CH * 16 * 4);

    auto blk = [](int n) { return (n + 255) / 256; };

    // -- weight conversion --
    auto wc = [&](const float* W, _Float16* Wt, int Nc, int K) {
        wconv_kernel<<<blk(Nc * K), 256, 0, stream>>>(W, Wt, Nc, K);
    };
    wc(Wl3, wl3t, 128, 64);  wc(Wr3, wr3t, 128, 64);
    wc(Wl4, wl4t, 64, 128);  wc(Wr4, wr4t, 64, 128);
    wc(Wl1, wl1t, 128, 64);  wc(Wr1, wr1t, 128, 64);
    wc(Wl2, wl2t, 64, 128);  wc(Wr2, wr2t, 64, 128);
    wc(Wfc,  wfct,  512, 256);
    wc(Wfc1, wfc1t, 256, 512);
    wc(Wmu,  wmut,  16, 256);
    wc(Wstd, wstdt, 16, 256);

    build_h16_kernel<<<blk(N * 64), 256, 0, stream>>>(x, nd, h16, N);

    // -- WMMA GEMM launcher (2 M-tiles per wave) --
    auto gemm = [&](const _Float16* A, const _Float16* Wt, const float* bias,
                    float* oF, _Float16* oH, int M, int K, int Nc, int relu) {
        int mg = (M + 31) / 32;
        if (Nc % 64 == 0) {
            int waves = mg * (Nc / 64);
            wmma_gemm_kernel<4><<<(waves + 7) / 8, 256, 0, stream>>>(A, Wt, bias, oF, oH, M, K, Nc, relu);
        } else {
            int waves = mg * (Nc / 16);
            wmma_gemm_kernel<1><<<(waves + 7) / 8, 256, 0, stream>>>(A, Wt, bias, oF, oH, M, K, Nc, relu);
        }
    };

    // -- one GATv2 layer --
    auto gat_layer = [&](const _Float16* in16, int K, int H, int C,
                         const _Float16* Wlt, const _Float16* Wrt,
                         const float* att, const float* bias,
                         const int* esrc, const int* edst, _Float16* out16) {
        int HC = H * C;
        gemm(in16, Wlt, nullptr, xl, nullptr, N, K, HC, 0);
        gemm(in16, Wrt, nullptr, xr, nullptr, N, K, HC, 0);
        fill_f32_kernel<<<blk(N * H), 256, 0, stream>>>(mbuf, -3.4e38f, N * H);
        fill_f32_kernel<<<blk(N * H), 256, 0, stream>>>(den, 0.0f, N * H);
        fill_f32_kernel<<<blk(N * HC), 256, 0, stream>>>(agg, 0.0f, N * HC);
        if (H == 4) {
            gat_score_kernel<4, 32><<<blk(Etot * 4), 256, 0, stream>>>(xl, xr, att, esrc, edst, E, Etot, score, mbuf);
            gat_expsum_kernel<4><<<blk(Etot * 4), 256, 0, stream>>>(score, mbuf, esrc, edst, E, Etot, den);
            gat_aggr_kernel<4, 32><<<blk(Etot * 4), 256, 0, stream>>>(score, den, xl, esrc, edst, E, Etot, agg);
        } else {
            gat_score_kernel<1, 64><<<blk(Etot), 256, 0, stream>>>(xl, xr, att, esrc, edst, E, Etot, score, mbuf);
            gat_expsum_kernel<1><<<blk(Etot), 256, 0, stream>>>(score, mbuf, esrc, edst, E, Etot, den);
            gat_aggr_kernel<1, 64><<<blk(Etot), 256, 0, stream>>>(score, den, xl, esrc, edst, E, Etot, agg);
        }
        gat_finish_kernel<<<blk(N * HC), 256, 0, stream>>>(agg, bias, out16, N, HC);
    };

    const int* fwd_src = ei;        // edge_index[0]
    const int* fwd_dst = ei + E;    // edge_index[1]
    // reverse layers: src=edge_index[1], dst=edge_index[0]
    gat_layer(h16,    64,  4, 32, wl3t, wr3t, att3, b3, fwd_dst, fwd_src, xb3_16);
    gat_layer(xb3_16, 128, 1, 64, wl4t, wr4t, att4, b4, fwd_dst, fwd_src, xb4_16);
    gat_layer(xb4_16, 64,  4, 32, wl1t, wr1t, att1, b1, fwd_src, fwd_dst, xf1_16);
    gat_layer(xf1_16, 128, 1, 64, wl2t, wr2t, att2, b2, fwd_src, fwd_dst, xf2_16);

    concat_s_kernel<<<blk(N * 128), 256, 0, stream>>>(xf2_16, xb4_16, s16, N);

    zero_scalar_kernel<<<1, 1, 0, stream>>>(out_scalar);

    // -- chunked edge MLP --
    for (int e0 = 0; e0 < E; e0 += CH) {
        int cm = (E - e0 < CH) ? (E - e0) : CH;
        gather_se_kernel<<<blk(cm * 256), 256, 0, stream>>>(s16, ei, E, e0, cm, se16);
        gemm(se16,  wfct,  bfc,  nullptr, h1_16, cm, 256, 512, 1);
        gemm(h1_16, wfc1t, bfc1, nullptr, h2_16, cm, 512, 256, 1);
        gemm(h2_16, wmut,  bmu,  muB, nullptr, cm, 256, 16, 0);
        gemm(h2_16, wstdt, bstd, spB, nullptr, cm, 256, 16, 0);
        mlp_head_kernel<<<blk(cm * NGOODS), 256, 0, stream>>>(muB, spB, eps, e0, cm, out, out_scalar);
    }
}
